// GraphLayer_44659069944167
// MI455X (gfx1250) — matrix operand order repaired and checked by reading, not compiled, not run on hardware
//
#include <hip/hip_runtime.h>
#include <hip/hip_bf16.h>

// ---------------------------------------------------------------------------
// GraphLayer (3x masked relational attention + output projection) for MI455X
// (gfx1250). All matmuls via v_wmma_f32_16x16x32_bf16 (wave32).
//
// Pipeline:
//   k0: node f32 -> bf16 (row-major + transposed [B,H,N])
//   k1: W0/W1/W2 and params f32 -> bf16 transposed (so every WMMA B-operand
//       is k-contiguous per lane -> 2x16B global loads per fragment)
//   k2: P_h = node @ W_h               (bf16, WMMA)
//   k3: fused masked attention: S = P @ node^T (WMMA, scores in LDS),
//       row softmax (wave32 shuffle reduce), O = prob @ node (WMMA, A-frags
//       from LDS via ds_load_b128), written directly as concat [B,N,3H] bf16
//   k4: out = w_cat @ params           (f32 output, WMMA)
// ---------------------------------------------------------------------------

typedef __attribute__((ext_vector_type(16))) __bf16 v16bf;
typedef __attribute__((ext_vector_type(8)))  __bf16 v8bf;
typedef __attribute__((ext_vector_type(8)))  float  v8f;

#define B_   8
#define N_   1024
#define H_   768
#define H3_  2304

static __device__ __forceinline__ unsigned short f2bf(float f) {
    unsigned u = __float_as_uint(f);
    u += 0x7FFFu + ((u >> 16) & 1u);          // round-to-nearest-even
    return (unsigned short)(u >> 16);
}

static __device__ __forceinline__ v16bf cat8(v8bf lo, v8bf hi) {
    v16bf r;
#pragma unroll
    for (int i = 0; i < 8; ++i) { r[i] = lo[i]; r[i + 8] = hi[i]; }
    return r;
}

// A-matrix 16x32 bf16 fragment (ISA 7.12.2): lanes 0-15 hold K 0-7 (V0-3) and
// K 16-23 (V4-7) of row M=lane; lanes 16-31 hold K 8-15 / 24-31 of M=lane-16.
// Source is row-major [row][k] with 'stride' elements per row.
static __device__ __forceinline__ v16bf load_a_frag(const unsigned short* p,
                                                    int stride, int row0,
                                                    int k0, int lane) {
    const int m     = lane & 15;
    const int abase = (lane >> 4) << 3;       // 0 or 8
    const unsigned short* q = p + (size_t)(row0 + m) * stride + k0 + abase;
    v8bf lo = *reinterpret_cast<const v8bf*>(q);        // K = abase..abase+7
    v8bf hi = *reinterpret_cast<const v8bf*>(q + 16);   // K = abase+16..+23
    return cat8(lo, hi);
}

// B-matrix 32x16 bf16 fragment: lane holds column N=lane&15, K matching A
// (lanes 0-15: K 0-7,16-23; lanes 16-31: K 8-15,24-31). Source is "n-major" layout
// [n][k] (i.e. the transposed matrix), k contiguous.
static __device__ __forceinline__ v16bf load_b_frag(const unsigned short* p,
                                                    int stride, int col0,
                                                    int k0, int lane) {
    const int n     = lane & 15;
    const int bbase = (lane >> 4) << 3;       // 0 or 8
    const unsigned short* q = p + (size_t)(col0 + n) * stride + k0 + bbase;
    v8bf lo = *reinterpret_cast<const v8bf*>(q);
    v8bf hi = *reinterpret_cast<const v8bf*>(q + 16);
    return cat8(lo, hi);
}

static __device__ __forceinline__ v8f wmma_bf16(v16bf a, v16bf b, v8f c) {
    return __builtin_amdgcn_wmma_f32_16x16x32_bf16(
        /*neg_a=*/false, a, /*neg_b=*/false, b,
        /*c_mod=*/(short)0, c, /*reuse_a=*/false, /*reuse_b=*/false);
}

// C/D 16x16 f32 layout: VGPR r -> row r (lanes 0-15) / row 8+r (lanes 16-31),
// column = lane & 15.
static __device__ __forceinline__ void store_c_bf16(unsigned short* p,
                                                    int stride, int row0,
                                                    int col0, int lane, v8f c) {
    const int rbase = (lane >> 4) << 3;
    const int n     = lane & 15;
#pragma unroll
    for (int r = 0; r < 8; ++r)
        p[(size_t)(row0 + rbase + r) * stride + col0 + n] = f2bf(c[r]);
}

static __device__ __forceinline__ void store_c_f32(float* p, int stride,
                                                   int row0, int col0,
                                                   int lane, v8f c) {
    const int rbase = (lane >> 4) << 3;
    const int n     = lane & 15;
#pragma unroll
    for (int r = 0; r < 8; ++r)
        p[(size_t)(row0 + rbase + r) * stride + col0 + n] = c[r];
}

// ---------------------------------------------------------------------------
// k0: node f32 -> bf16 row-major [B,N,H] and transposed [B,H,N]
// ---------------------------------------------------------------------------
__global__ __launch_bounds__(256) void convert_node_kernel(
    const float* __restrict__ node, unsigned short* __restrict__ nodeBF,
    unsigned short* __restrict__ nodeT) {
    long idx = (long)blockIdx.x * blockDim.x + threadIdx.x;
    const long total = (long)B_ * N_ * H_;
    if (idx >= total) return;
    int b = (int)(idx / ((long)N_ * H_));
    int rem = (int)(idx - (long)b * N_ * H_);
    int n = rem / H_;
    int h = rem - n * H_;
    unsigned short v = f2bf(node[idx]);
    nodeBF[idx] = v;
    nodeT[((size_t)b * H_ + h) * N_ + n] = v;
}

// ---------------------------------------------------------------------------
// k1: generic f32 [rows][cols] -> bf16 transposed [cols][rows]
// ---------------------------------------------------------------------------
__global__ __launch_bounds__(256) void transpose_bf16_kernel(
    const float* __restrict__ src, unsigned short* __restrict__ dst,
    int rows, int cols) {
    long idx = (long)blockIdx.x * blockDim.x + threadIdx.x;
    if (idx >= (long)rows * cols) return;
    int r = (int)(idx / cols);
    int c = (int)(idx - (long)r * cols);
    dst[(size_t)c * rows + r] = f2bf(src[idx]);
}

// ---------------------------------------------------------------------------
// k2: P[b,h] = node[b] @ W_h   ([1024x768] @ [768x768], bf16 out)
// grid.x = B*3*(N/16); 8 waves/block, each wave 6 of the 48 column tiles.
// ---------------------------------------------------------------------------
__global__ __launch_bounds__(256) void proj_kernel(
    const unsigned short* __restrict__ nodeBF,
    const unsigned short* __restrict__ WT,   // [3][768][768], WT[h][j][k]
    unsigned short* __restrict__ Pbuf) {     // [B][3][1024][768]
    const int x    = blockIdx.x;
    const int b    = x / 192;
    const int rem  = x - b * 192;
    const int head = rem >> 6;
    const int rt   = rem & 63;
    const int w    = threadIdx.x >> 5;
    const int lane = threadIdx.x & 31;

    const unsigned short* A  = nodeBF + (size_t)b * N_ * H_;
    const unsigned short* Bp = WT + (size_t)head * H_ * H_;
    unsigned short* out = Pbuf + (size_t)(b * 3 + head) * N_ * H_;
    const int row0 = rt * 16;

    for (int i = 0; i < 6; ++i) {
        const int col0 = (w + 8 * i) * 16;
        v8f c = {};
        for (int kk = 0; kk < H_; kk += 32) {
            v16bf a  = load_a_frag(A, H_, row0, kk, lane);
            v16bf bf = load_b_frag(Bp, H_, col0, kk, lane);
            c = wmma_bf16(a, bf, c);
        }
        store_c_bf16(out, H_, row0, col0, lane, c);
    }
}

// ---------------------------------------------------------------------------
// k3: fused masked attention for one (b, head, 16-row tile).
// grid.x = B*3*(N/16); 256 threads; 96 KB dynamic LDS.
// ---------------------------------------------------------------------------
__global__ __launch_bounds__(256) void attend_kernel(
    const unsigned short* __restrict__ Pbuf,    // [B][3][N][H]
    const unsigned short* __restrict__ nodeBF,  // [B][N][H]
    const unsigned short* __restrict__ nodeT,   // [B][H][N]
    const int* __restrict__ adjR, const int* __restrict__ adjO,
    const int* __restrict__ adjU,
    unsigned short* __restrict__ wcat) {        // [B][N][3H]
    extern __shared__ unsigned char smem[];
    float*          S  = reinterpret_cast<float*>(smem);            // 16x1024 f32
    unsigned short* Pp = reinterpret_cast<unsigned short*>(smem + 16 * N_ * 4);

    const int x    = blockIdx.x;
    const int b    = x / 192;
    const int rem  = x - b * 192;
    const int head = rem >> 6;
    const int rt   = rem & 63;
    const int w    = threadIdx.x >> 5;
    const int lane = threadIdx.x & 31;
    const int row0 = rt * 16;

    const unsigned short* A  = Pbuf + (size_t)(b * 3 + head) * N_ * H_;
    const unsigned short* Nb = nodeBF + (size_t)b * N_ * H_;
    const unsigned short* Nt = nodeT + (size_t)b * H_ * N_;
    const int* adj = (head == 0) ? adjR : (head == 1) ? adjO : adjU;

    // ---- Phase A: S = P_tile @ node^T, masked, into LDS -------------------
    for (int i = 0; i < 8; ++i) {
        const int col0 = (w + 8 * i) * 16;
        v8f c = {};
        for (int kk = 0; kk < H_; kk += 32) {
            v16bf a  = load_a_frag(A, H_, row0, kk, lane);
            v16bf bf = load_b_frag(Nb, H_, col0, kk, lane);  // node^T access
            c = wmma_bf16(a, bf, c);
        }
        const int rbase = (lane >> 4) << 3;
        const int col   = col0 + (lane & 15);
#pragma unroll
        for (int r = 0; r < 8; ++r) {
            const int rloc = rbase + r;
            const int amask =
                adj[((size_t)b * N_ + row0 + rloc) * N_ + col];
            S[rloc * N_ + col] = (amask == 1) ? c[r] : -1.0e7f;
        }
    }
    __syncthreads();

    // ---- Phase B: row softmax (wave32 shuffle reductions) -----------------
    for (int rr = 0; rr < 2; ++rr) {
        const int row = w * 2 + rr;
        float v[32];
        float mx = -3.4e38f;
#pragma unroll
        for (int i = 0; i < 32; ++i) {
            v[i] = S[row * N_ + i * 32 + lane];
            mx = fmaxf(mx, v[i]);
        }
        for (int m = 16; m; m >>= 1) mx = fmaxf(mx, __shfl_xor(mx, m, 32));
        float sum = 0.0f;
#pragma unroll
        for (int i = 0; i < 32; ++i) {
            float e = __expf(v[i] - mx);
            v[i] = e;
            sum += e;
        }
        for (int m = 16; m; m >>= 1) sum += __shfl_xor(sum, m, 32);
        const float inv = 1.0f / sum;
#pragma unroll
        for (int i = 0; i < 32; ++i)
            Pp[row * N_ + i * 32 + lane] = f2bf(v[i] * inv);
    }
    __syncthreads();

    // ---- Phase C: O = prob @ node, written as concat slice ----------------
    unsigned short* out = wcat + (size_t)head * H_;  // column offset in [.,3H]
    for (int i = 0; i < 6; ++i) {
        const int col0 = (w + 8 * i) * 16;
        v8f c = {};
        for (int kk = 0; kk < N_; kk += 32) {
            v16bf a  = load_a_frag(Pp, N_, 0, kk, lane);   // ds_load_b128 x2
            v16bf bf = load_b_frag(Nt, N_, col0, kk, lane);
            c = wmma_bf16(a, bf, c);
        }
        store_c_bf16(out, H3_, (b * N_ + row0), col0, lane, c);
    }
}

// ---------------------------------------------------------------------------
// k4: out = w_cat @ params   ([8192x2304] @ [2304x768], f32 out)
// grid.x = (B*N)/16; 8 waves, each wave 6 of 48 column tiles; K = 72 steps.
// ---------------------------------------------------------------------------
__global__ __launch_bounds__(256) void final_kernel(
    const unsigned short* __restrict__ wcat,     // [8192][2304]
    const unsigned short* __restrict__ paramsT,  // [768][2304]
    float* __restrict__ out) {                   // [8192][768]
    const int rt   = blockIdx.x;
    const int w    = threadIdx.x >> 5;
    const int lane = threadIdx.x & 31;
    const int row0 = rt * 16;

    for (int i = 0; i < 6; ++i) {
        const int col0 = (w + 8 * i) * 16;
        v8f c = {};
        for (int kk = 0; kk < H3_; kk += 32) {
            v16bf a  = load_a_frag(wcat, H3_, row0, kk, lane);
            v16bf bf = load_b_frag(paramsT, H3_, col0, kk, lane);
            c = wmma_bf16(a, bf, c);
        }
        store_c_f32(out, H_, row0, col0, lane, c);
    }
}

// ---------------------------------------------------------------------------
// Workspace layout (bytes, all 256B aligned). Total ~102.8 MB.
// ---------------------------------------------------------------------------
#define OFF_NODEBF  0u
#define OFF_NODET   12582912u   // + B*N*H*2
#define OFF_WT      25165824u   // + B*N*H*2
#define OFF_PARAMST 28704768u   // + 3*H*H*2
#define OFF_P       32243712u   // + H*3H*2
#define OFF_WCAT    69992448u   // + 3*B*N*H*2

extern "C" void kernel_launch(void* const* d_in, const int* in_sizes, int n_in,
                              void* d_out, int out_size, void* d_ws,
                              size_t ws_size, hipStream_t stream) {
    (void)in_sizes; (void)n_in; (void)out_size; (void)ws_size;

    const float* node    = (const float*)d_in[0];
    const int*   adjR    = (const int*)d_in[1];
    const int*   adjO    = (const int*)d_in[2];
    const int*   adjU    = (const int*)d_in[3];
    const float* Wr      = (const float*)d_in[4];
    const float* Wo      = (const float*)d_in[5];
    const float* Wu      = (const float*)d_in[6];
    const float* params  = (const float*)d_in[7];
    float*       out     = (float*)d_out;

    char* ws = (char*)d_ws;
    unsigned short* nodeBF  = (unsigned short*)(ws + OFF_NODEBF);
    unsigned short* nodeT   = (unsigned short*)(ws + OFF_NODET);
    unsigned short* WT      = (unsigned short*)(ws + OFF_WT);
    unsigned short* paramsT = (unsigned short*)(ws + OFF_PARAMST);
    unsigned short* Pbuf    = (unsigned short*)(ws + OFF_P);
    unsigned short* wcat    = (unsigned short*)(ws + OFF_WCAT);

    // k0: node -> bf16 (+ transposed copy)
    {
        const long total = (long)B_ * N_ * H_;
        dim3 g((unsigned)((total + 255) / 256));
        convert_node_kernel<<<g, 256, 0, stream>>>(node, nodeBF, nodeT);
    }
    // k1: pre-transpose W0..2 and params to bf16
    {
        dim3 gw((H_ * H_ + 255) / 256);
        transpose_bf16_kernel<<<gw, 256, 0, stream>>>(Wr, WT + 0 * H_ * H_, H_, H_);
        transpose_bf16_kernel<<<gw, 256, 0, stream>>>(Wo, WT + 1 * H_ * H_, H_, H_);
        transpose_bf16_kernel<<<gw, 256, 0, stream>>>(Wu, WT + 2 * H_ * H_, H_, H_);
        dim3 gp((H3_ * H_ + 255) / 256);
        transpose_bf16_kernel<<<gp, 256, 0, stream>>>(params, paramsT, H3_, H_);
    }
    // k2: projections P_h = node @ W_h
    proj_kernel<<<B_ * 3 * (N_ / 16), 256, 0, stream>>>(nodeBF, WT, Pbuf);

    // k3: fused masked attention (96 KB dynamic LDS: 64K scores + 32K probs)
    attend_kernel<<<B_ * 3 * (N_ / 16), 256, 16 * N_ * 4 + 16 * N_ * 2, stream>>>(
        Pbuf, nodeBF, nodeT, adjR, adjO, adjU, wcat);

    // k4: output projection
    final_kernel<<<(B_ * N_) / 16, 256, 0, stream>>>(wcat, paramsT, out);
}
